// MACAM_858993459837
// MI455X (gfx1250) — matrix operand order repaired and checked by reading, not compiled, not run on hardware
//
#include <hip/hip_runtime.h>

typedef __attribute__((ext_vector_type(8)))  float  v8f;
typedef __attribute__((ext_vector_type(16))) __bf16 v16bf;

union FragU {
    unsigned int u[8];
    v16bf v;
};

__device__ __forceinline__ unsigned int pack_bf2(float a, float b) {
    unsigned int ua = __float_as_uint(a), ub = __float_as_uint(b);
    ua = (ua + 0x7FFFu + ((ua >> 16) & 1u)) >> 16;
    ub = (ub + 0x7FFFu + ((ub >> 16) & 1u)) >> 16;
    return (ua & 0xFFFFu) | (ub << 16);
}

__device__ __forceinline__ unsigned short f2bf(float x) {
    unsigned int u = __float_as_uint(x);
    return (unsigned short)((u + 0x7FFFu + ((u >> 16) & 1u)) >> 16);
}

// B=16, L=24, D=256, Q=256, C=256, H=W=128, HW=16384
#define NB 16
#define NL 24
#define ND 256
#define NC 256
#define NHW 16384

// ---------------------------------------------------------------------------
// Kernel A1: per-batch small stuff. scores -> argmax -> bg ; k -> Kc, kb
// one block per batch, 256 threads
// ---------------------------------------------------------------------------
__global__ __launch_bounds__(256) void kA1(
    const float* __restrict__ wsrc, const float* __restrict__ wtgt,
    const float* __restrict__ fck_w, const float* __restrict__ fck_b,
    const float* __restrict__ fc_w,  const float* __restrict__ fc_b,
    const float* __restrict__ conv0_w, const float* __restrict__ conv0_b,
    float* __restrict__ gBG, float* __restrict__ gKc, float* __restrict__ gKB)
{
    __shared__ float sWS[NL * ND];   // w_source[b]
    __shared__ float sWT[NL * ND];   // w_target[b], later reused as k
    __shared__ float sSC[NL * NL];
    __shared__ float sCOL[NL];
    __shared__ int   sIDX[NL];

    const int t = threadIdx.x;
    const int b = blockIdx.x;
    const float* wsb = wsrc + (size_t)b * NL * ND;
    const float* wtb = wtgt + (size_t)b * NL * ND;

    for (int e = t; e < NL * ND; e += 256) { sWS[e] = wsb[e]; sWT[e] = wtb[e]; }
    __syncthreads();

    // scores[l][m] = <ws[l], wt[m]>
    for (int e = t; e < NL * NL; e += 256) {
        int l = e / NL, m = e % NL;
        float s = 0.f;
        for (int d = 0; d < ND; ++d) s += sWS[l * ND + d] * sWT[m * ND + d];
        sSC[e] = s;
    }
    __syncthreads();

    if (t < NL) {               // column sums over l
        float s = 0.f;
        for (int l = 0; l < NL; ++l) s += sSC[l * NL + t];
        sCOL[t] = s;
    }
    __syncthreads();

    if (t < NL) {               // idx[l] = argmax_m (colsum[m] - scores[l][m]), first max
        float best = -3.4e38f; int bi = 0;
        for (int m = 0; m < NL; ++m) {
            float v = sCOL[m] - sSC[t * NL + m];
            if (v > best) { best = v; bi = m; }
        }
        sIDX[t] = bi;
    }
    __syncthreads();

    // bg[l][e] = fc_b[e] + <wt[idx[l]], fc_w[e]>
    for (int e = t; e < NL * 512; e += 256) {
        int l = e >> 9, ee = e & 511;
        const float* wrow = &sWT[sIDX[l] * ND];
        const float* fw = &fc_w[(size_t)ee * ND];
        float s = fc_b[ee];
        for (int d = 0; d < ND; ++d) s += wrow[d] * fw[d];
        gBG[((size_t)b * NL + l) * 512 + ee] = s;
    }
    __syncthreads();

    // k[l][q] = fck_b[q] + <ws[l], fck_w[q]>  (written into sWT, now free)
    for (int e = t; e < NL * ND; e += 256) {
        int l = e >> 8, q = e & 255;
        const float* fw = &fck_w[(size_t)q * ND];
        float s = fck_b[q];
        for (int d = 0; d < ND; ++d) s += sWS[l * ND + d] * fw[d];
        sWT[l * ND + q] = s;
    }
    __syncthreads();

    // Kc[l][c] = sum_q k[l][q] * conv0_w[q][c]
    for (int e = t; e < NL * NC; e += 256) {
        int l = e >> 8, c = e & 255;
        float s = 0.f;
        for (int q = 0; q < 256; ++q) s += sWT[l * ND + q] * conv0_w[(size_t)q * NC + c];
        gKc[((size_t)b * NL + l) * NC + c] = s;
    }
    if (t < NL) {               // kb[l] = sum_q conv0_b[q]*k[l][q]
        float s = 0.f;
        for (int q = 0; q < 256; ++q) s += conv0_b[q] * sWT[t * ND + q];
        gKB[b * NL + t] = s;
    }
}

// ---------------------------------------------------------------------------
// Kernel A2: Gt[b][e][c] = sum_l Kc[b][l][c] * bg[b][l][e]  (bf16, transposed)
//            v[b][e]     = sum_l kb[b][l]   * bg[b][l][e]
// grid (32 e-tiles, 16 batches), 256 threads; thread = c, 16 e's per thread
// ---------------------------------------------------------------------------
__global__ __launch_bounds__(256) void kA2(
    const float* __restrict__ gKc, const float* __restrict__ gBG,
    const float* __restrict__ gKB,
    unsigned short* __restrict__ Gt, float* __restrict__ gV)
{
    __shared__ float sKc[NL * NC];
    __shared__ float sbg[NL * 16];
    __shared__ float skb[NL];

    const int t = threadIdx.x;
    const int e0 = blockIdx.x * 16;
    const int b = blockIdx.y;

    for (int e = t; e < NL * NC; e += 256) sKc[e] = gKc[(size_t)b * NL * NC + e];
    if (t < NL * 16) {
        int l = t >> 4, j = t & 15;
        sbg[t] = gBG[((size_t)b * NL + l) * 512 + e0 + j];
    }
    if (t < NL) skb[t] = gKB[b * NL + t];
    __syncthreads();

    const int c = t;
    float acc[16];
#pragma unroll
    for (int j = 0; j < 16; ++j) acc[j] = 0.f;
    for (int l = 0; l < NL; ++l) {
        float kc = sKc[l * NC + c];
#pragma unroll
        for (int j = 0; j < 16; ++j) acc[j] += kc * sbg[l * 16 + j];
    }
#pragma unroll
    for (int j = 0; j < 16; ++j)
        Gt[((size_t)b * 512 + e0 + j) * NC + c] = f2bf(acc[j]);

    if (t < 16) {
        float s = 0.f;
        for (int l = 0; l < NL; ++l) s += skb[l] * sbg[l * 16 + t];
        gV[b * 512 + e0 + t] = s;
    }
}

// ---------------------------------------------------------------------------
// Kernel B: instance-norm stats -> fused scale/shift per (b,c)
// grid (256 c, 16 b), 256 threads
// ---------------------------------------------------------------------------
__global__ __launch_bounds__(256) void kB(
    const float* __restrict__ h, const float* __restrict__ in_w,
    const float* __restrict__ in_b,
    float* __restrict__ gScale, float* __restrict__ gShift)
{
    const int c = blockIdx.x, b = blockIdx.y, t = threadIdx.x;
    const float* p = h + ((size_t)(b * NC + c)) * NHW;
    float s = 0.f, q = 0.f;
    for (int i = t; i < NHW; i += 256) { float v = p[i]; s += v; q += v * v; }
    __shared__ float rs[256], rq[256];
    rs[t] = s; rq[t] = q;
    __syncthreads();
    for (int off = 128; off > 0; off >>= 1) {
        if (t < off) { rs[t] += rs[t + off]; rq[t] += rq[t + off]; }
        __syncthreads();
    }
    if (t == 0) {
        float mean = rs[0] * (1.f / NHW);
        float var = rq[0] * (1.f / NHW) - mean * mean;
        float inv = rsqrtf(var + 1e-5f);
        float sc = inv * in_w[c];
        gScale[b * NC + c] = sc;
        gShift[b * NC + c] = in_b[c] - mean * sc;
    }
}

// ---------------------------------------------------------------------------
// Kernel C: WMMA GEMM + fused instance-norm modulation.
// Per batch: D[c', hw] pairs (beta,gamma) = Gt(c' x c) * h(c x hw) + v
// out = (h*scale+shift)*gamma_map + beta_map
// Block tile: M=64 (c'), N=128 (hw), K chunked by 32. 8 waves (2x4),
// each wave: 2 m-tiles x 2 n-tiles x {beta,gamma} = 8 v8f accumulators.
// ---------------------------------------------------------------------------
__global__ __launch_bounds__(256) void kC(
    const float* __restrict__ h, const unsigned int* __restrict__ gtu,
    const float* __restrict__ gV, const float* __restrict__ gScale,
    const float* __restrict__ gShift, float* __restrict__ out)
{
    __shared__ unsigned int Ald[2 * 64 * 18];  // [mat][m][kpair], pitch 18 dwords
    __shared__ unsigned int Bld[128 * 18];     // [n][kpair],      pitch 18 dwords
    __shared__ float sS[64], sH[64], sVB[64], sVG[64];

    const int t = threadIdx.x;
    const int b = blockIdx.z;
    const int c0 = blockIdx.y * 64;
    const int hw0 = blockIdx.x * 128;

    if (t < 64) {
        sS[t]  = gScale[b * NC + c0 + t];
        sH[t]  = gShift[b * NC + c0 + t];
        sVB[t] = gV[b * 512 + c0 + t];
        sVG[t] = gV[b * 512 + 256 + c0 + t];
    }

    const int lane = t & 31, half = lane >> 4, lm = lane & 15;
    const int wid = t >> 5;
    const int wm = wid & 1;        // 0..1 -> m-tiles {2wm, 2wm+1}
    const int wn = wid >> 1;       // 0..3 -> n-tiles {2wn, 2wn+1}

    const v8f zero8 = {0.f, 0.f, 0.f, 0.f, 0.f, 0.f, 0.f, 0.f};
    v8f acc[2][2][2];
#pragma unroll
    for (int mi = 0; mi < 2; ++mi)
#pragma unroll
        for (int ni = 0; ni < 2; ++ni) { acc[mi][ni][0] = zero8; acc[mi][ni][1] = zero8; }

    const float* hb = h + (size_t)b * NC * NHW;

    for (int kc = 0; kc < 8; ++kc) {
        const int k0 = kc * 32;
        // --- A tiles: Gt rows [beta: c0..c0+63][gamma: 256+c0..], 16 bf16 pairs ---
#pragma unroll
        for (int i = 0; i < 8; ++i) {
            int e = t + 256 * i;           // 2048 dwords
            int mat = e >> 10, rem = e & 1023;
            int m = rem >> 4, kp = rem & 15;
            int row = b * 512 + mat * 256 + c0 + m;
            Ald[mat * 1152 + m * 18 + kp] = gtu[(size_t)row * 128 + kc * 16 + kp];
        }
        // --- B tile: h[k0..k0+31][hw0..hw0+127] -> bf16 pairs along c ---
#pragma unroll
        for (int i = 0; i < 8; ++i) {
            int e = t + 256 * i;           // 2048 dwords
            int n = e & 127, kp = e >> 7;
            size_t base = (size_t)(k0 + 2 * kp) * NHW + hw0 + n;
            Bld[n * 18 + kp] = pack_bf2(hb[base], hb[base + NHW]);
        }
        __syncthreads();

        // --- fragments ---
        FragU afb[2], afg[2], bfr[2];
#pragma unroll
        for (int mi = 0; mi < 2; ++mi) {
            int m = (wm * 2 + mi) * 16 + lm;
            int b0 = m * 18, b1 = 1152 + m * 18;
#pragma unroll
            for (int j = 0; j < 4; ++j) {
                afb[mi].u[j]     = Ald[b0 + 4 * half + j];
                afb[mi].u[4 + j] = Ald[b0 + 8 + 4 * half + j];
                afg[mi].u[j]     = Ald[b1 + 4 * half + j];
                afg[mi].u[4 + j] = Ald[b1 + 8 + 4 * half + j];
            }
        }
#pragma unroll
        for (int ni = 0; ni < 2; ++ni) {
            int n = (wn * 2 + ni) * 16 + lm;
#pragma unroll
            for (int j = 0; j < 8; ++j) bfr[ni].u[j] = Bld[n * 18 + 4 * half + j + (j >= 4 ? 4 : 0)];
        }

        // --- WMMA ---
#pragma unroll
        for (int mi = 0; mi < 2; ++mi)
#pragma unroll
            for (int ni = 0; ni < 2; ++ni) {
                acc[mi][ni][0] = __builtin_amdgcn_wmma_f32_16x16x32_bf16(
                    false, afb[mi].v, false, bfr[ni].v, (short)0, acc[mi][ni][0], false, false);
                acc[mi][ni][1] = __builtin_amdgcn_wmma_f32_16x16x32_bf16(
                    false, afg[mi].v, false, bfr[ni].v, (short)0, acc[mi][ni][1], false, false);
            }
        __syncthreads();
    }

    // --- epilogue: out = (h*scale+shift)*(gamma+vg) + (beta+vb) ---
#pragma unroll
    for (int mi = 0; mi < 2; ++mi) {
#pragma unroll
        for (int ni = 0; ni < 2; ++ni) {
            int mtl = (wm * 2 + mi) * 16;
            int ntl = (wn * 2 + ni) * 16;
            int hw = hw0 + ntl + lm;
#pragma unroll
            for (int i = 0; i < 8; ++i) {
                int cpl = mtl + i + 8 * half;               // D: M = i + 8*half
                size_t gi = ((size_t)(b * NC + c0 + cpl)) * NHW + hw;
                float hv = h[gi];
                out[gi] = (hv * sS[cpl] + sH[cpl]) * (acc[mi][ni][1][i] + sVG[cpl])
                        + (acc[mi][ni][0][i] + sVB[cpl]);
            }
        }
    }
}

// ---------------------------------------------------------------------------
extern "C" void kernel_launch(void* const* d_in, const int* in_sizes, int n_in,
                              void* d_out, int out_size, void* d_ws, size_t ws_size,
                              hipStream_t stream) {
    (void)in_sizes; (void)n_in; (void)out_size; (void)ws_size;
    const float* w_source = (const float*)d_in[0];
    const float* w_target = (const float*)d_in[1];
    const float* h        = (const float*)d_in[2];
    const float* conv0_w  = (const float*)d_in[3];
    const float* conv0_b  = (const float*)d_in[4];
    const float* fck_w    = (const float*)d_in[5];
    const float* fck_b    = (const float*)d_in[6];
    const float* fc_w     = (const float*)d_in[7];
    const float* fc_b     = (const float*)d_in[8];
    const float* in_w     = (const float*)d_in[9];
    const float* in_b     = (const float*)d_in[10];

    char* ws = (char*)d_ws;
    size_t off = 0;
    float* gBG = (float*)(ws + off); off += (size_t)NB * NL * 512 * 4;
    float* gKc = (float*)(ws + off); off += (size_t)NB * NL * NC * 4;
    float* gKB = (float*)(ws + off); off += (size_t)NB * NL * 4;
    float* gV  = (float*)(ws + off); off += (size_t)NB * 512 * 4;
    float* gScale = (float*)(ws + off); off += (size_t)NB * NC * 4;
    float* gShift = (float*)(ws + off); off += (size_t)NB * NC * 4;
    unsigned short* Gt = (unsigned short*)(ws + off); off += (size_t)NB * 512 * NC * 2;

    kA1<<<dim3(NB), 256, 0, stream>>>(w_source, w_target, fck_w, fck_b, fc_w, fc_b,
                                      conv0_w, conv0_b, gBG, gKc, gKB);
    kA2<<<dim3(32, NB), 256, 0, stream>>>(gKc, gBG, gKB, Gt, gV);
    kB<<<dim3(NC, NB), 256, 0, stream>>>(h, in_w, in_b, gScale, gShift);
    kC<<<dim3(NHW / 128, NC / 64, NB), 256, 0, stream>>>(
        h, (const unsigned int*)Gt, gV, gScale, gShift, (float*)d_out);
}
